// SNNReadoutModuleSoftmaxForRecord_20538533609757
// MI455X (gfx1250) — matrix operand order repaired and checked by reading, not compiled, not run on hardware
//
#include <hip/hip_runtime.h>

// ---------------------------------------------------------------------------
// SNN readout: h2 = x @ W  (65536 x 256 x 1024, bf16 WMMA, f32 accum)
// then per-batch leaky-integrate scan accumulating softmax(mem) over 511 steps.
// Target: MI455X / gfx1250, wave32, v_wmma_f32_16x16x32_bf16.
//
// Data movement plan (23.3 TB/s HBM, 192 MB L2):
//  - x (256 MB) read once, NON-TEMPORAL (don't evict h2 from L2)
//  - W packed once to bf16 B-fragment layout (512 KB, L2-resident)
//  - h2 (64 MB) written RT -> stays in L2 -> scan loads are L2-latency,
//    hidden by a depth-8 register prefetch pipeline.
// ---------------------------------------------------------------------------

typedef __attribute__((ext_vector_type(16))) __bf16 v16bf;
typedef __attribute__((ext_vector_type(8)))  float  v8f;
typedef __attribute__((ext_vector_type(4)))  float  f32x4;

#define B_DIM 128
#define T_DIM 512
#define H_DIM 1024
#define O_DIM 256

// ALPHA = exp(-0.1) in double, rounded to f32 (matches jax weak-typed scalar
// semantics); ONE_MINUS = float(1.0 - exp(-0.1)).
#define ALPHA_F     0.90483741803595957f
#define ONE_MINUS_F 0.09516258196404043f

// ---------------------------------------------------------------------------
// Kernel 0: pack W (H x O, f32 row-major) into per-lane bf16 B-fragments.
// Fragment (kt, nt): 32 lanes x 16 bf16.  lane: half = lane>>4, n = nt*16 +
// (lane&15); element e holds W[kt*32 + half*8 + e + (e>=8 ? 8 : 0)][n].
// ---------------------------------------------------------------------------
__global__ __launch_bounds__(256) void pack_w_kernel(
    const float* __restrict__ w, __bf16* __restrict__ wb) {
  const int idx  = blockIdx.x * 256 + threadIdx.x;   // 0 .. 16383
  const int lane = idx & 31;
  const int nt   = (idx >> 5) & 15;                  // 0..15  (N tile)
  const int kt   = idx >> 9;                         // 0..31  (K tile)
  const int half = lane >> 4;
  const int n    = nt * 16 + (lane & 15);
  const int kb   = kt * 32 + half * 8;
  v16bf v;
#pragma unroll
  for (int e = 0; e < 16; ++e) {
    v[e] = (__bf16)w[(size_t)(kb + e + (e >= 8 ? 8 : 0)) * O_DIM + n];
  }
  *((v16bf*)wb + idx) = v;
}

// ---------------------------------------------------------------------------
// Kernel 1: GEMM  h2[row][n] = sum_k x[row][k] * W[k][n]
// Block: 256 threads = 8 waves; block tile = 64(M) x 256(N).
// Wave w: mhalf = w>>2 -> 32 rows (two 16-row A fragments), nq = w&3 -> 64
// cols (4 B tiles).  8 accumulators/wave; each B fragment feeds 2 WMMAs.
// K loop: 1024 / 32 = 32 iterations; 8 x v_wmma_f32_16x16x32_bf16 per iter.
// x loads are non-temporal (streaming; preserve h2 in L2 for the scan).
// ---------------------------------------------------------------------------
__global__ __launch_bounds__(256) void gemm_bf16_kernel(
    const float* __restrict__ x, const __bf16* __restrict__ wb,
    float* __restrict__ h2) {
  const int tid   = threadIdx.x;
  const int lane  = tid & 31;
  const int w     = tid >> 5;
  const int half  = lane >> 4;
  const int l15   = lane & 15;
  const int mhalf = w >> 2;      // 0..1 -> 32-row half of the 64-row block tile
  const int nq    = w & 3;       // 0..3 -> 64-col strip
  const int row0  = blockIdx.x * 64 + mhalf * 32;

  // A: lane reads rows (row0 + l15) and (row0 + 16 + l15).
  const float* ap = x + (size_t)(row0 + l15) * H_DIM;

  v8f acc[2][4];
#pragma unroll
  for (int r = 0; r < 2; ++r)
#pragma unroll
    for (int j = 0; j < 4; ++j) acc[r][j] = (v8f){0, 0, 0, 0, 0, 0, 0, 0};

  const v16bf* bbase = (const v16bf*)wb;

  for (int kt = 0; kt < 32; ++kt) {
    const int k0 = kt * 32;
    v16bf a[2];
#pragma unroll
    for (int r = 0; r < 2; ++r) {
      const float* arp = ap + (size_t)(r * 16) * H_DIM;
      // elems 0..7  -> K = k0 + half*8 + 0..7
      // elems 8..15 -> K = k0 + 16 + half*8 + 0..7
      const f32x4 f0 =
          __builtin_nontemporal_load((const f32x4*)(arp + k0 + half * 8));
      const f32x4 f1 =
          __builtin_nontemporal_load((const f32x4*)(arp + k0 + half * 8 + 4));
      const f32x4 f2 = __builtin_nontemporal_load(
          (const f32x4*)(arp + k0 + 16 + half * 8));
      const f32x4 f3 = __builtin_nontemporal_load(
          (const f32x4*)(arp + k0 + 16 + half * 8 + 4));
      a[r][0]  = (__bf16)f0.x; a[r][1]  = (__bf16)f0.y;
      a[r][2]  = (__bf16)f0.z; a[r][3]  = (__bf16)f0.w;
      a[r][4]  = (__bf16)f1.x; a[r][5]  = (__bf16)f1.y;
      a[r][6]  = (__bf16)f1.z; a[r][7]  = (__bf16)f1.w;
      a[r][8]  = (__bf16)f2.x; a[r][9]  = (__bf16)f2.y;
      a[r][10] = (__bf16)f2.z; a[r][11] = (__bf16)f2.w;
      a[r][12] = (__bf16)f3.x; a[r][13] = (__bf16)f3.y;
      a[r][14] = (__bf16)f3.z; a[r][15] = (__bf16)f3.w;
    }

    const v16bf* bp = bbase + ((size_t)kt * 16 + nq * 4) * 32 + lane;
#pragma unroll
    for (int j = 0; j < 4; ++j) {
      const v16bf bfrag = bp[j * 32];
      acc[0][j] = __builtin_amdgcn_wmma_f32_16x16x32_bf16(
          false, a[0], false, bfrag, (short)0, acc[0][j], false, false);
      acc[1][j] = __builtin_amdgcn_wmma_f32_16x16x32_bf16(
          false, a[1], false, bfrag, (short)0, acc[1][j], false, false);
    }
  }

  // D layout: N = lane&15, VGPR v holds M = v + 8*half.
  const int n0 = nq * 64;
#pragma unroll
  for (int r = 0; r < 2; ++r) {
#pragma unroll
    for (int j = 0; j < 4; ++j) {
#pragma unroll
      for (int v = 0; v < 8; ++v) {
        h2[(size_t)(row0 + r * 16 + 8 * half + v) * O_DIM + n0 + j * 16 +
           l15] = acc[r][j][v];
      }
    }
  }
}

// ---------------------------------------------------------------------------
// Kernel 2: per-batch scan.  One wave (32 lanes) per b; lane owns 8 outputs
// o = lane*8 .. lane*8+7 (coalesced 1KB/wave/step).  Depth-8 register
// prefetch pipeline hides the (L2-class) load latency of the serial t loop.
// ---------------------------------------------------------------------------
__global__ __launch_bounds__(32) void scan_softmax_kernel(
    const float* __restrict__ h2, float* __restrict__ out) {
  const int b    = blockIdx.x;
  const int lane = threadIdx.x;
  const float* base = h2 + (size_t)b * T_DIM * O_DIM + lane * 8;

  float mem[8], acc[8];
#pragma unroll
  for (int j = 0; j < 8; ++j) { mem[j] = 0.0f; acc[j] = 0.0f; }

  constexpr int STEPS = T_DIM - 1;  // 511
  constexpr int DEPTH = 8;
  f32x4 buf[DEPTH][2];
#pragma unroll
  for (int i = 0; i < DEPTH; ++i) {
    buf[i][0] = *(const f32x4*)(base + (size_t)i * O_DIM);
    buf[i][1] = *(const f32x4*)(base + (size_t)i * O_DIM + 4);
  }

#pragma unroll 8
  for (int t = 0; t < STEPS; ++t) {
    const int s = t & (DEPTH - 1);
    const f32x4 a0 = buf[s][0];
    const f32x4 a1 = buf[s][1];
    const int tn = t + DEPTH;
    if (tn < STEPS) {
      buf[s][0] = *(const f32x4*)(base + (size_t)tn * O_DIM);
      buf[s][1] = *(const f32x4*)(base + (size_t)tn * O_DIM + 4);
    }
    const float h[8] = {a0.x, a0.y, a0.z, a0.w, a1.x, a1.y, a1.z, a1.w};

    float m = -3.402823466e38f;
#pragma unroll
    for (int j = 0; j < 8; ++j) {
      mem[j] = ALPHA_F * mem[j] + ONE_MINUS_F * h[j];
      m = fmaxf(m, mem[j]);
    }
#pragma unroll
    for (int off = 16; off > 0; off >>= 1)
      m = fmaxf(m, __shfl_xor(m, off, 32));

    float e[8];
    float ssum = 0.0f;
#pragma unroll
    for (int j = 0; j < 8; ++j) {
      e[j] = __expf(mem[j] - m);
      ssum += e[j];
    }
#pragma unroll
    for (int off = 16; off > 0; off >>= 1)
      ssum += __shfl_xor(ssum, off, 32);

    const float inv = 1.0f / ssum;
#pragma unroll
    for (int j = 0; j < 8; ++j) acc[j] += e[j] * inv;
  }

  float* op = out + (size_t)b * O_DIM + lane * 8;
#pragma unroll
  for (int j = 0; j < 8; ++j) op[j] = acc[j];
}

// ---------------------------------------------------------------------------
extern "C" void kernel_launch(void* const* d_in, const int* in_sizes, int n_in,
                              void* d_out, int out_size, void* d_ws,
                              size_t ws_size, hipStream_t stream) {
  (void)in_sizes; (void)n_in; (void)out_size; (void)ws_size;
  const float* x = (const float*)d_in[0];        // [128,512,1024] f32
  const float* w = (const float*)d_in[1];        // [1024,256]     f32
  float* out = (float*)d_out;                    // [128,256]      f32

  const size_t h2_bytes = (size_t)B_DIM * T_DIM * O_DIM * sizeof(float); // 64MB
  float*  h2 = (float*)d_ws;
  __bf16* wb = (__bf16*)((char*)d_ws + h2_bytes);                        // 512KB

  // 1) pack W -> bf16 B-fragments (64 blocks x 256 threads)
  pack_w_kernel<<<64, 256, 0, stream>>>(w, wb);
  // 2) GEMM: 65536 rows / 64 per block = 1024 blocks x 256 threads
  gemm_bf16_kernel<<<1024, 256, 0, stream>>>(x, wb, h2);
  // 3) scan: one wave per batch row
  scan_softmax_kernel<<<B_DIM, 32, 0, stream>>>(h2, out);
}
